// MatrixMPowerSeriesLayer_7713761264376
// MI455X (gfx1250) — matrix operand order repaired and checked by reading, not compiled, not run on hardware
//
#include <hip/hip_runtime.h>

// ---------------------------------------------------------------------------
// Degree-7 complex matrix polynomial via Horner on gfx1250 (MI455X):
//   S = C7; for j = 6..0: S = C_j + S @ X      (B=64 batches, N=512, complex)
//
// v2 pipeline (used when ws_size permits):
//   - one-time pre-pass: X -> bf16 hi/lo planes, column-major (LDS-tile ready)
//                        C7 -> bf16 hi/lo planes, row-major
//   - 7 GEMM steps: A (S) and B (X) as split bf16, streamed into DOUBLE-
//     BUFFERED LDS with GLOBAL_LOAD_ASYNC_TO_LDS_B128; stage s+1's fill
//     overlaps stage s's 48 WMMAs via s_wait_asynccnt <= 12 (in-order compl.);
//     12x v_wmma_f32_16x16x32_bf16 per 16x16 tile (hi/lo split complex mul);
//     epilogue adds C_j, writes S in split form (f32 only on final step).
//   - d_out doubles as one S ping-pong buffer until the final step.
// Fallback (small ws): round-1 kernel with in-kernel f32->bf16 conversion.
// ---------------------------------------------------------------------------

#define NMAT 512
#define PLANE (512 * 512)
#define BM 128
#define BN 64
#define BK 32
#define SAK 40                      // padded LDS K-stride (ushorts), 80B rows
#define SBK 40

typedef __attribute__((ext_vector_type(16))) __bf16          v16bf;
typedef __attribute__((ext_vector_type(8)))  __bf16          v8bf;
typedef __attribute__((ext_vector_type(16))) unsigned short  v16u;
typedef __attribute__((ext_vector_type(8)))  float           v8f;
typedef int v4i __attribute__((vector_size(4 * sizeof(int))));

union V16 { v16bf v; v8bf h[2]; };

__device__ __forceinline__ unsigned short f2bh(float f) {
  unsigned int u = __float_as_uint(f);
  u += 0x7FFFu + ((u >> 16) & 1u);           // RNE
  return (unsigned short)(u >> 16);
}
__device__ __forceinline__ float bh2f(unsigned short h) {
  return __uint_as_float(((unsigned int)h) << 16);
}
__device__ __forceinline__ void split2(float f, unsigned short &hi, unsigned short &lo) {
  hi = f2bh(f);
  lo = f2bh(f - bh2f(hi));                    // ~2^-17 combined error
}
__device__ __forceinline__ v16bf negbf(v16bf a) {
  v16u t;
  __builtin_memcpy(&t, &a, sizeof(t));
  t ^= (unsigned short)0x8000;
  v16bf r;
  __builtin_memcpy(&r, &t, sizeof(r));
  return r;
}

// ---- CDNA5 async global->LDS copy (16B per lane) + asynccnt waits ---------
#if __has_builtin(__builtin_amdgcn_global_load_async_to_lds_b128)
#define ASYNC_CP16(g, l)                                                     \
  __builtin_amdgcn_global_load_async_to_lds_b128(                            \
      (v4i __attribute__((address_space(1)))*)(uintptr_t)(g),                \
      (v4i __attribute__((address_space(3)))*)(unsigned)(uintptr_t)(l), 0, 0)
#else
#define ASYNC_CP16(g, l)                                                     \
  asm volatile("global_load_async_to_lds_b128 %0, %1, off"                   \
               :: "v"((unsigned)(uintptr_t)(l)),                             \
                  "v"((unsigned long long)(uintptr_t)(g)) : "memory")
#endif

#if __has_builtin(__builtin_amdgcn_s_wait_asynccnt)
#define WAIT_ASYNC(n) __builtin_amdgcn_s_wait_asynccnt(n)
#else
#define WAIT_ASYNC(n) asm volatile("s_wait_asynccnt %0" :: "i"(n) : "memory")
#endif

#define WMMA_BF16(A, Bm, C) \
  __builtin_amdgcn_wmma_f32_16x16x32_bf16(false, (A), false, (Bm), (short)0, (C), false, false)

// ---- shared MMA core: fragments from LDS + 48 WMMAs per K-stage -----------
__device__ __forceinline__ void stage_mma(
    const unsigned short (*sA)[BM * SAK], const unsigned short (*sB)[BN * SBK],
    int lane, int wm, int wn, v8f accR[2][2], v8f accI[2][2])
{
  const int rsel = lane & 15;
  const int aKb  = (lane >= 16) ? 8 : 0;      // A: K runs {0-7,16-23}/{8-15,24-31}
  const int bKb  = (lane >= 16) ? 8 : 0;      // B: K runs {0-7,16-23}/{8-15,24-31}

  v16bf af[2][6];                             // Arh,Arl,Aih,Ail,-Aih,-Ail
#pragma unroll
  for (int mi = 0; mi < 2; ++mi) {
    int row = wm * 32 + mi * 16 + rsel;
#pragma unroll
    for (int q = 0; q < 4; ++q) {
      V16 u;
      u.h[0] = *(const v8bf*)&sA[q][row * SAK + aKb];
      u.h[1] = *(const v8bf*)&sA[q][row * SAK + aKb + 16];
      af[mi][q] = u.v;
    }
    af[mi][4] = negbf(af[mi][2]);
    af[mi][5] = negbf(af[mi][3]);
  }
  v16bf bfr[2][4];                            // Xrh,Xrl,Xih,Xil
#pragma unroll
  for (int ni = 0; ni < 2; ++ni) {
    int col = wn * 32 + ni * 16 + rsel;
#pragma unroll
    for (int q = 0; q < 4; ++q) {
      V16 u;
      u.h[0] = *(const v8bf*)&sB[q][col * SBK + bKb];
      u.h[1] = *(const v8bf*)&sB[q][col * SBK + bKb + 16];
      bfr[ni][q] = u.v;
    }
  }

#pragma unroll
  for (int mi = 0; mi < 2; ++mi)
#pragma unroll
    for (int ni = 0; ni < 2; ++ni) {
      v8f r = accR[mi][ni];
      r = WMMA_BF16(af[mi][0], bfr[ni][0], r);   //  Arh*Xrh
      r = WMMA_BF16(af[mi][0], bfr[ni][1], r);   //  Arh*Xrl
      r = WMMA_BF16(af[mi][1], bfr[ni][0], r);   //  Arl*Xrh
      r = WMMA_BF16(af[mi][4], bfr[ni][2], r);   // -Aih*Xih
      r = WMMA_BF16(af[mi][4], bfr[ni][3], r);   // -Aih*Xil
      r = WMMA_BF16(af[mi][5], bfr[ni][2], r);   // -Ail*Xih
      accR[mi][ni] = r;
      v8f im = accI[mi][ni];
      im = WMMA_BF16(af[mi][0], bfr[ni][2], im); //  Arh*Xih
      im = WMMA_BF16(af[mi][0], bfr[ni][3], im); //  Arh*Xil
      im = WMMA_BF16(af[mi][1], bfr[ni][2], im); //  Arl*Xih
      im = WMMA_BF16(af[mi][2], bfr[ni][0], im); //  Aih*Xrh
      im = WMMA_BF16(af[mi][2], bfr[ni][1], im); //  Aih*Xrl
      im = WMMA_BF16(af[mi][3], bfr[ni][0], im); //  Ail*Xrh
      accI[mi][ni] = im;
    }
}

// ---- async fill of one K-stage (12 x 16B copies per thread) ---------------
__device__ __forceinline__ void issue_stage(
    const unsigned short* Ab, const unsigned short* Xb,
    int m0, int n0, int kk,
    unsigned short (*dA)[BM * SAK], unsigned short (*dB)[BN * SBK], int tid)
{
  // A tile: 4 planes x 128 rows x 4 chunks(16B) = 2048 copies, 8/thread
#pragma unroll
  for (int t = 0; t < 8; ++t) {
    int idx = tid + t * 256;
    int p   = idx >> 9;
    int rem = idx & 511;
    int m   = rem >> 2;
    int kc  = (rem & 3) << 3;                // 0,8,16,24 elements
    const unsigned short* g = Ab + (size_t)p * PLANE + (size_t)(m0 + m) * NMAT + kk + kc;
    ASYNC_CP16(g, &dA[p][m * SAK + kc]);
    if (kk + BK < NMAT) __builtin_prefetch(g + BK, 0, 1);
  }
  // B tile: 4 planes x 64 cols x 4 chunks = 1024 copies, 4/thread
#pragma unroll
  for (int t = 0; t < 4; ++t) {
    int idx = tid + t * 256;
    int p   = idx >> 8;
    int rem = idx & 255;
    int n   = rem >> 2;
    int kc  = (rem & 3) << 3;
    const unsigned short* g = Xb + (size_t)p * PLANE + (size_t)(n0 + n) * NMAT + kk + kc;
    ASYNC_CP16(g, &dB[p][n * SBK + kc]);
    if (kk + BK < NMAT) __builtin_prefetch(g + BK, 0, 1);
  }
}

// ===========================================================================
// Pre-pass 1: X (B,2,N,N) f32 -> 4 bf16 planes per batch, COLUMN-major [n][k]
// ===========================================================================
__global__ __launch_bounds__(256) void split_x_kernel(
    const float* __restrict__ X, unsigned short* __restrict__ XB)
{
  __shared__ float tile[32][33];
  const int bp = blockIdx.z;                 // b*2 + plane(0=r,1=i)
  const int b  = bp >> 1, P = bp & 1;
  const int k0 = blockIdx.y * 32;
  const int n0 = blockIdx.x * 32;
  const int tx = threadIdx.x & 31;
  const int ty = (threadIdx.x >> 5) & 7;

  const float* src = X + (size_t)bp * PLANE;
#pragma unroll
  for (int i = 0; i < 32; i += 8)
    tile[ty + i][tx] = src[(size_t)(k0 + ty + i) * NMAT + n0 + tx];
  __syncthreads();

  unsigned short* dh = XB + (size_t)b * 4 * PLANE + (size_t)(P * 2) * PLANE;
  unsigned short* dl = dh + PLANE;
#pragma unroll
  for (int i = 0; i < 32; i += 8) {
    float v = tile[tx][ty + i];              // element (k=k0+tx, n=n0+ty+i)
    unsigned short h, l;
    split2(v, h, l);
    size_t o = (size_t)(n0 + ty + i) * NMAT + k0 + tx;
    dh[o] = h;
    dl[o] = l;
  }
}

// ===========================================================================
// Pre-pass 2: C7 (2,N,N) f32 -> 4 bf16 planes, ROW-major (A-side layout)
// ===========================================================================
__global__ __launch_bounds__(256) void split_c7_kernel(
    const float* __restrict__ C7, unsigned short* __restrict__ C7s)
{
  int idx = blockIdx.x * 256 + threadIdx.x;  // 0 .. 2*PLANE-1
  int P = idx >= PLANE;
  int o = idx - P * PLANE;
  unsigned short h, l;
  split2(C7[idx], h, l);
  C7s[(size_t)(P * 2) * PLANE + o]     = h;
  C7s[(size_t)(P * 2 + 1) * PLANE + o] = l;
}

// ===========================================================================
// v2 Horner step: double-buffered async-LDS pipeline, split-bf16 end to end
// ===========================================================================
template <bool FINAL>
__global__ __launch_bounds__(256) void horner_step_v2(
    const unsigned short* __restrict__ Asplit, long long aStride, // ushorts/batch
    const unsigned short* __restrict__ Xsplit,                    // col-major split
    const float* __restrict__ Cadd,                               // C_j f32
    float* __restrict__ OutF,                                     // FINAL only
    unsigned short* __restrict__ OutS)                            // !FINAL only
{
  __shared__ unsigned short sA[2][4][BM * SAK];   // 80 KB
  __shared__ unsigned short sB[2][4][BN * SBK];   // 40 KB

  const int tid  = threadIdx.x;
  const int lane = tid & 31;
  const int wave = tid >> 5;
  const int wm   = wave >> 1;
  const int wn   = wave & 1;

  const int b  = blockIdx.z;
  const int m0 = blockIdx.y * BM;
  const int n0 = blockIdx.x * BN;

  const unsigned short* Ab = Asplit + (long long)b * aStride;
  const unsigned short* Xb = Xsplit + (size_t)b * 4 * PLANE;

  v8f accR[2][2], accI[2][2];
  {
    v8f z = {};
    for (int i = 0; i < 2; ++i)
      for (int j = 0; j < 2; ++j) { accR[i][j] = z; accI[i][j] = z; }
  }

  issue_stage(Ab, Xb, m0, n0, 0, sA[0], sB[0], tid);
  for (int s = 0; s < NMAT / BK; ++s) {
    const int kk = s * BK;
    if (kk + BK < NMAT) {
      issue_stage(Ab, Xb, m0, n0, kk + BK, sA[(s + 1) & 1], sB[(s + 1) & 1], tid);
      WAIT_ASYNC(12);                 // stage s done (in-order), s+1 in flight
    } else {
      WAIT_ASYNC(0);
    }
    __syncthreads();
    stage_mma(sA[s & 1], sB[s & 1], lane, wm, wn, accR, accI);
    __syncthreads();                  // stage s buffer safe for reuse at s+2
  }

  // ---- epilogue: S = acc + C_j ----
  const float* cr = Cadd;
  const float* ci = Cadd + PLANE;
  const int colL = lane & 15;
  const int rowH = (lane >> 4) * 8;

  if (FINAL) {
    float* outR = OutF + (size_t)b * 2 * PLANE;
    float* outI = outR + PLANE;
#pragma unroll
    for (int mi = 0; mi < 2; ++mi)
#pragma unroll
      for (int ni = 0; ni < 2; ++ni) {
        int gmB = m0 + wm * 32 + mi * 16 + rowH;
        int gn  = n0 + wn * 32 + ni * 16 + colL;
#pragma unroll
        for (int r = 0; r < 8; ++r) {
          size_t o = (size_t)(gmB + r) * NMAT + gn;
          outR[o] = accR[mi][ni][r] + cr[o];
          outI[o] = accI[mi][ni][r] + ci[o];
        }
      }
  } else {
    unsigned short* srh = OutS + (size_t)b * 4 * PLANE;
    unsigned short* srl = srh + PLANE;
    unsigned short* sih = srl + PLANE;
    unsigned short* sil = sih + PLANE;
#pragma unroll
    for (int mi = 0; mi < 2; ++mi)
#pragma unroll
      for (int ni = 0; ni < 2; ++ni) {
        int gmB = m0 + wm * 32 + mi * 16 + rowH;
        int gn  = n0 + wn * 32 + ni * 16 + colL;
#pragma unroll
        for (int r = 0; r < 8; ++r) {
          size_t o = (size_t)(gmB + r) * NMAT + gn;
          unsigned short h, l;
          split2(accR[mi][ni][r] + cr[o], h, l);
          srh[o] = h; srl[o] = l;
          split2(accI[mi][ni][r] + ci[o], h, l);
          sih[o] = h; sil[o] = l;
        }
      }
  }
}

// ===========================================================================
// Fallback (round-1): f32 operands, in-kernel split, VGPR-staged LDS fill
// ===========================================================================
__global__ __launch_bounds__(256) void horner_step_v1(
    const float* __restrict__ Abase, long long aStride,
    const float* __restrict__ Xbase,
    const float* __restrict__ Cadd,
    float* __restrict__ Out)
{
  __shared__ unsigned short sA[4][BM * SAK];
  __shared__ unsigned short sB[4][BN * SBK];

  const int tid  = threadIdx.x;
  const int lane = tid & 31;
  const int wave = tid >> 5;
  const int wm   = wave >> 1;
  const int wn   = wave & 1;

  const int b  = blockIdx.z;
  const int m0 = blockIdx.y * BM;
  const int n0 = blockIdx.x * BN;

  const float* Ar = Abase + (long long)b * aStride;
  const float* Ai = Ar + PLANE;
  const float* Xr = Xbase + (size_t)b * 2 * PLANE;
  const float* Xi = Xr + PLANE;

  v8f accR[2][2], accI[2][2];
  {
    v8f z = {};
    for (int i = 0; i < 2; ++i)
      for (int j = 0; j < 2; ++j) { accR[i][j] = z; accI[i][j] = z; }
  }

  for (int kk = 0; kk < NMAT; kk += BK) {
    __syncthreads();
#pragma unroll
    for (int t = 0; t < 8; ++t) {
      int idx = tid + t * 256;
      int mat = idx >> 10;
      int rem = idx & 1023;
      int m   = rem >> 3;
      int kq  = (rem & 7) << 2;
      const float* src = (mat ? Ai : Ar) + (size_t)(m0 + m) * NMAT + kk + kq;
      float4 v = *(const float4*)src;
      unsigned short* H = &sA[mat * 2][m * SAK + kq];
      unsigned short* L = &sA[mat * 2 + 1][m * SAK + kq];
      split2(v.x, H[0], L[0]);
      split2(v.y, H[1], L[1]);
      split2(v.z, H[2], L[2]);
      split2(v.w, H[3], L[3]);
    }
#pragma unroll
    for (int t = 0; t < 4; ++t) {
      int idx = tid + t * 256;
      int mat = idx >> 9;
      int rem = idx & 511;
      int kl  = rem >> 4;
      int n4  = (rem & 15) << 2;
      const float* src = (mat ? Xi : Xr) + (size_t)(kk + kl) * NMAT + n0 + n4;
      float4 v = *(const float4*)src;
      float vv[4] = {v.x, v.y, v.z, v.w};
#pragma unroll
      for (int c = 0; c < 4; ++c) {
        unsigned short h, l;
        split2(vv[c], h, l);
        int off = (n4 + c) * SBK + kl;
        sB[mat * 2][off]     = h;
        sB[mat * 2 + 1][off] = l;
      }
    }
    __syncthreads();
    stage_mma(sA, sB, lane, wm, wn, accR, accI);
  }

  const float* cr = Cadd;
  const float* ci = Cadd + PLANE;
  float* outR = Out + (size_t)b * 2 * PLANE;
  float* outI = outR + PLANE;
  const int colL = lane & 15;
  const int rowH = (lane >> 4) * 8;
#pragma unroll
  for (int mi = 0; mi < 2; ++mi)
#pragma unroll
    for (int ni = 0; ni < 2; ++ni) {
      int gmB = m0 + wm * 32 + mi * 16 + rowH;
      int gn  = n0 + wn * 32 + ni * 16 + colL;
#pragma unroll
      for (int r = 0; r < 8; ++r) {
        size_t o = (size_t)(gmB + r) * NMAT + gn;
        outR[o] = accR[mi][ni][r] + cr[o];
        outI[o] = accI[mi][ni][r] + ci[o];
      }
    }
}

// ===========================================================================
extern "C" void kernel_launch(void* const* d_in, const int* in_sizes, int n_in,
                              void* d_out, int out_size, void* d_ws, size_t ws_size,
                              hipStream_t stream) {
  (void)in_sizes; (void)n_in; (void)out_size;
  const float* x      = (const float*)d_in[0];   // (64,2,512,512)
  const float* coeffs = (const float*)d_in[1];   // (8,2,512,512)
  float* O = (float*)d_out;

  const long long plane2   = 2LL * PLANE;
  const size_t    bigElems = 64ull * 4 * PLANE;          // ushorts per split buf
  const size_t    needWS   = (2 * bigElems + 4ull * PLANE) * sizeof(unsigned short);

  dim3 grid(NMAT / BN, NMAT / BM, 64);
  dim3 block(256);

  if (ws_size >= needWS) {
    unsigned short* XB  = (unsigned short*)d_ws;         // X split, col-major
    unsigned short* S0  = XB + bigElems;                 // S ping (ws)
    unsigned short* S1  = (unsigned short*)d_out;        // S pong (reuses out)
    unsigned short* C7s = S0 + bigElems;                 // C7 split, row-major

    split_x_kernel<<<dim3(16, 16, 128), block, 0, stream>>>(x, XB);
    split_c7_kernel<<<dim3(2 * PLANE / 256), block, 0, stream>>>(
        coeffs + 7 * plane2, C7s);

    // t writes: S1,S0,S1,S0,S1,S0, then final f32 -> d_out (reads S0). j = 6-t.
    const unsigned short* Asrc = C7s;
    long long aStride = 0;
    for (int t = 0; t < 6; ++t) {
      unsigned short* dst = (t & 1) ? S0 : S1;
      horner_step_v2<false><<<grid, block, 0, stream>>>(
          Asrc, aStride, XB, coeffs + (long long)(6 - t) * plane2, nullptr, dst);
      Asrc = dst;
      aStride = (long long)4 * PLANE;
    }
    horner_step_v2<true><<<grid, block, 0, stream>>>(
        S0, (long long)4 * PLANE, XB, coeffs /* C_0 */, O, nullptr);
  } else {
    // Fallback: f32 ping-pong between d_out and d_ws (7 steps end in d_out).
    float* W = (float*)d_ws;
    const float* Asrc = coeffs + 7 * plane2;
    long long aStride = 0;
    for (int t = 0; t < 7; ++t) {
      float* dst = (t & 1) ? W : O;
      horner_step_v1<<<grid, block, 0, stream>>>(
          Asrc, aStride, x, coeffs + (long long)(6 - t) * plane2, dst);
      Asrc = dst;
      aStride = plane2;
    }
  }
}
